// MixFormerBlock_14998025798118
// MI455X (gfx1250) — matrix operand order repaired and checked by reading, not compiled, not run on hardware
//
#include <hip/hip_runtime.h>
#include <hip/hip_bf16.h>
#include <math.h>

typedef __attribute__((ext_vector_type(16))) _Float16 v16h;
typedef __attribute__((ext_vector_type(8)))  float    v8f;

#define B_  16
#define T_  1024
#define H_  16
#define HD_ 128
#define D_  2048
#define HF_ 8192
#define HQ_ 512
#define NU_ 4
#define CH_ 8
#define EPS_ 1e-8f

#define MT 128
#define NT 128
#define KT 32

// ---- CDNA5 async global->LDS copy (probe-verified signature) ----
#if __has_builtin(__builtin_amdgcn_global_load_async_to_lds_b128) && \
    __has_builtin(__builtin_amdgcn_s_wait_asynccnt)
#define HAVE_ASYNC_LDS 1
typedef __attribute__((__vector_size__(16))) int vi4;
#define ASYNC_CP16(gsrc, ldst)                                          \
  __builtin_amdgcn_global_load_async_to_lds_b128((vi4*)(gsrc),          \
                                                 (vi4*)(ldst), 0, 0)
#define ASYNC_WAIT() __builtin_amdgcn_s_wait_asynccnt(0)
#else
#define HAVE_ASYNC_LDS 0
#define ASYNC_CP16(gsrc, ldst) \
  do { *(uint4*)(ldst) = *(const uint4*)(gsrc); } while (0)
#define ASYNC_WAIT() do {} while (0)
#endif

static __device__ __forceinline__ int lane_id() { return threadIdx.x & 31; }

static __device__ __forceinline__ float wave_sum(float v) {
#pragma unroll
  for (int off = 16; off > 0; off >>= 1) v += __shfl_xor(v, off, 32);
  return v;
}

// ---- WMMA fragment helpers (CDNA5 wave32 layouts, CDNA5 ISA) ----

// A fragment: 16x32 f16 (MxK) from LDS, row-major, stride ld (elements).
static __device__ __forceinline__ v16h load_a_lds(const _Float16* s, int ld) {
  const int lane = lane_id();
  const int m  = lane & 15;
  const int kb = (lane < 16) ? 0 : 16;
  v16h a;
#pragma unroll
  for (int i = 0; i < 8; ++i) a[i]     = s[m * ld + kb + i];
#pragma unroll
  for (int i = 0; i < 8; ++i) a[8 + i] = s[m * ld + 8 + kb + i];
  return a;
}

// B fragment from pre-swizzled LDS: fragment-lane's 16 f16 contiguous (32B).
static __device__ __forceinline__ v16h load_b_swz(const _Float16* base, int sub) {
  return *(const v16h*)(base + (size_t)(sub * 32 + (threadIdx.x & 31)) * 16);
}

// Scatter one f32 element (k-row r, n-col c) into the swizzled B buffer:
// fragment lane = (r&16) | (c&15), element = r&15.
static __device__ __forceinline__ _Float16* swz_dst(_Float16* base, int r, int c) {
  return base + (size_t)(((c >> 4) * 32 + (r & 16) + (c & 15)) * 16 + (r & 15));
}

static __device__ __forceinline__ v8f wmma_f16(v16h a, v16h b, v8f c) {
  return __builtin_amdgcn_wmma_f32_16x16x32_f16(false, a, false, b, (short)0, c,
                                                false, false);
}

// ---------------------------------------------------------------------------
// Kernel 1: sn = rmsnorm(seq_repr, sf_norm_w) -> f16   [B*T, D]
// ---------------------------------------------------------------------------
__global__ __launch_bounds__(256) void k_seq_rmsnorm(const float* __restrict__ x,
                                                     const float* __restrict__ w,
                                                     _Float16* __restrict__ xn) {
  const int row = blockIdx.x;
  const float* xr = x + (size_t)row * D_;
  __shared__ float red[256];
  float ss = 0.f;
#pragma unroll
  for (int i = 0; i < D_ / 256; ++i) {
    float v = xr[threadIdx.x + i * 256];
    ss += v * v;
  }
  red[threadIdx.x] = ss;
  __syncthreads();
  for (int s = 128; s > 0; s >>= 1) {
    if (threadIdx.x < s) red[threadIdx.x] += red[threadIdx.x + s];
    __syncthreads();
  }
  const float rstd = rsqrtf(red[0] / (float)D_ + EPS_);
  _Float16* outr = xn + (size_t)row * D_;
#pragma unroll
  for (int i = 0; i < D_ / 256; ++i) {
    int idx = threadIdx.x + i * 256;
    outr[idx] = (_Float16)(xr[idx] * rstd * w[idx]);
  }
}

// ---------------------------------------------------------------------------
// Kernel 2: h = silu(sn@Wg + gb) * (sn@Wv + vb) -> f16  [16384, 8192]
// Double-buffered: async A-copy + B-convert for tile k+1 overlap WMMA on k.
// ---------------------------------------------------------------------------
__global__ __launch_bounds__(256) void k_ffn_gateval(
    const _Float16* __restrict__ xn, const float* __restrict__ wg,
    const float* __restrict__ gb, const float* __restrict__ wv,
    const float* __restrict__ vb, _Float16* __restrict__ hbuf) {
  const int n0 = blockIdx.x * NT;
  const int m0 = blockIdx.y * MT;
  __shared__ __align__(16) _Float16 At[2][MT][KT + 8];
  __shared__ __align__(32) _Float16 BgS[2][(NT / 16) * 32 * 16];
  __shared__ __align__(32) _Float16 BvS[2][(NT / 16) * 32 * 16];
  const int tid = threadIdx.x;
  const int w = tid >> 5, lane = tid & 31;
  const int wm = w >> 2, wn = w & 3;
  v8f accg[4][2] = {};
  v8f accv[4][2] = {};

  auto stage = [&](int k0, int p) {
#pragma unroll
    for (int s = tid; s < 512; s += 256) {
      int r = s >> 2, c = (s & 3) * 8;
      ASYNC_CP16(xn + (size_t)(m0 + r) * D_ + k0 + c, &At[p][r][c]);
    }
#pragma unroll
    for (int s = tid; s < 1024; s += 256) {
      int r = s >> 5, c = (s & 31) * 4;
      float4 g4 = *(const float4*)(wg + (size_t)(k0 + r) * HF_ + n0 + c);
      _Float16* dg = swz_dst(&BgS[p][0], r, c);
      dg[0] = (_Float16)g4.x; dg[16] = (_Float16)g4.y;
      dg[32] = (_Float16)g4.z; dg[48] = (_Float16)g4.w;
      float4 v4 = *(const float4*)(wv + (size_t)(k0 + r) * HF_ + n0 + c);
      _Float16* dv = swz_dst(&BvS[p][0], r, c);
      dv[0] = (_Float16)v4.x; dv[16] = (_Float16)v4.y;
      dv[32] = (_Float16)v4.z; dv[48] = (_Float16)v4.w;
    }
  };

  stage(0, 0);
  ASYNC_WAIT();
  __syncthreads();
  const int NK = D_ / KT;
  for (int k = 0; k < NK; ++k) {
    const int p = k & 1;
    if (k + 1 < NK) stage((k + 1) * KT, p ^ 1);

    v16h af[4];
#pragma unroll
    for (int mi = 0; mi < 4; ++mi)
      af[mi] = load_a_lds(&At[p][wm * 64 + mi * 16][0], KT + 8);
#pragma unroll
    for (int ni = 0; ni < 2; ++ni) {
      const int sub = wn * 2 + ni;
      v16h bgf = load_b_swz(&BgS[p][0], sub);
      v16h bvf = load_b_swz(&BvS[p][0], sub);
#pragma unroll
      for (int mi = 0; mi < 4; ++mi) {
        accg[mi][ni] = wmma_f16(af[mi], bgf, accg[mi][ni]);
        accv[mi][ni] = wmma_f16(af[mi], bvf, accv[mi][ni]);
      }
    }
    ASYNC_WAIT();
    __syncthreads();
  }

  const int col = lane & 15;
#pragma unroll
  for (int ni = 0; ni < 2; ++ni) {
    const int n = n0 + wn * 32 + ni * 16 + col;
    const float gbias = gb[n], vbias = vb[n];
#pragma unroll
    for (int mi = 0; mi < 4; ++mi) {
#pragma unroll
      for (int r = 0; r < 8; ++r) {
        int m = m0 + wm * 64 + mi * 16 + ((lane < 16) ? r : r + 8);
        float g = accg[mi][ni][r] + gbias;
        float v = accv[mi][ni][r] + vbias;
        float hv = (g / (1.f + expf(-g))) * v;  // silu(g)*v
        // NT store: hbuf is stream-once, keep L2 for resident weights
        __builtin_nontemporal_store((_Float16)hv, &hbuf[(size_t)m * HF_ + n]);
      }
    }
  }
}

// ---------------------------------------------------------------------------
// Kernel 3: seq_hidden = h @ Wo + ob + seq_repr -> f32  [16384, 2048]
// ---------------------------------------------------------------------------
__global__ __launch_bounds__(256) void k_ffn_out(
    const _Float16* __restrict__ hbuf, const float* __restrict__ wo,
    const float* __restrict__ bo, const float* __restrict__ seq,
    float* __restrict__ out) {
  const int n0 = blockIdx.x * NT;
  const int m0 = blockIdx.y * MT;
  __shared__ __align__(16) _Float16 At[2][MT][KT + 8];
  __shared__ __align__(32) _Float16 BoS[2][(NT / 16) * 32 * 16];
  const int tid = threadIdx.x;
  const int w = tid >> 5, lane = tid & 31;
  const int wm = w >> 2, wn = w & 3;
  v8f acc[4][2] = {};

  auto stage = [&](int k0, int p) {
#pragma unroll
    for (int s = tid; s < 512; s += 256) {
      int r = s >> 2, c = (s & 3) * 8;
      ASYNC_CP16(hbuf + (size_t)(m0 + r) * HF_ + k0 + c, &At[p][r][c]);
    }
#pragma unroll
    for (int s = tid; s < 1024; s += 256) {
      int r = s >> 5, c = (s & 31) * 4;
      float4 f = *(const float4*)(wo + (size_t)(k0 + r) * D_ + n0 + c);
      _Float16* db = swz_dst(&BoS[p][0], r, c);
      db[0] = (_Float16)f.x; db[16] = (_Float16)f.y;
      db[32] = (_Float16)f.z; db[48] = (_Float16)f.w;
    }
  };

  stage(0, 0);
  ASYNC_WAIT();
  __syncthreads();
  const int NK = HF_ / KT;
  for (int k = 0; k < NK; ++k) {
    const int p = k & 1;
    if (k + 1 < NK) stage((k + 1) * KT, p ^ 1);

    v16h af[4];
#pragma unroll
    for (int mi = 0; mi < 4; ++mi)
      af[mi] = load_a_lds(&At[p][wm * 64 + mi * 16][0], KT + 8);
#pragma unroll
    for (int ni = 0; ni < 2; ++ni) {
      v16h bf = load_b_swz(&BoS[p][0], wn * 2 + ni);
#pragma unroll
      for (int mi = 0; mi < 4; ++mi)
        acc[mi][ni] = wmma_f16(af[mi], bf, acc[mi][ni]);
    }
    ASYNC_WAIT();
    __syncthreads();
  }

  const int col = lane & 15;
#pragma unroll
  for (int ni = 0; ni < 2; ++ni) {
    const int n = n0 + wn * 32 + ni * 16 + col;
    const float bias = bo[n];
#pragma unroll
    for (int mi = 0; mi < 4; ++mi) {
#pragma unroll
      for (int r = 0; r < 8; ++r) {
        int m = m0 + wm * 64 + mi * 16 + ((lane < 16) ? r : r + 8);
        float r0 = __builtin_nontemporal_load(&seq[(size_t)m * D_ + n]);
        __builtin_nontemporal_store(acc[mi][ni][r] + bias + r0,
                                    &out[(size_t)m * D_ + n]);
      }
    }
  }
}

// ---------------------------------------------------------------------------
// Kernel 4: QueryMixer pre: p = mask*mix(rmsnorm(x)) + x  -> f32 [B,H,HD]
// ---------------------------------------------------------------------------
__global__ __launch_bounds__(256) void k_qm_mix(const float* __restrict__ xh,
                                               const float* __restrict__ nw,
                                               float* __restrict__ pws) {
  const int b = blockIdx.x;
  __shared__ float X[H_ * HD_];
  __shared__ float N[H_ * HD_];
  const int tid = threadIdx.x, w = tid >> 5, lane = tid & 31;
#pragma unroll
  for (int i = 0; i < (H_ * HD_) / 256; ++i) {
    int idx = tid + i * 256;
    X[idx] = xh[(size_t)b * H_ * HD_ + idx];
  }
  __syncthreads();
#pragma unroll
  for (int j = 0; j < 2; ++j) {  // wave w handles heads 2w, 2w+1
    int hh = w * 2 + j;
    float ss = 0.f;
#pragma unroll
    for (int i = 0; i < 4; ++i) {
      float v = X[hh * HD_ + lane + i * 32];
      ss += v * v;
    }
    ss = wave_sum(ss);
    float rstd = rsqrtf(ss / (float)HD_ + EPS_);
#pragma unroll
    for (int i = 0; i < 4; ++i) {
      int d = lane + i * 32;
      N[hh * HD_ + d] = X[hh * HD_ + d] * rstd * nw[d];
    }
  }
  __syncthreads();
#pragma unroll
  for (int i = 0; i < (H_ * HD_) / 256; ++i) {
    int idx = tid + i * 256;
    int hh = idx >> 7, d = idx & 127;
    int c = d >> 3, jj = d & 7;
    float mixed = N[c * HD_ + hh * CH_ + jj];      // transpose head<->chunk
    if (hh < NU_ && d >= NU_ * CH_) mixed = 0.f;   // UI decoupling mask
    pws[(size_t)b * H_ * HD_ + idx] = mixed + X[idx];
  }
}

// ---------------------------------------------------------------------------
// Kernel 5 (shared): out = per_head_swiglu(rmsnorm(p, normw)) + p
// ---------------------------------------------------------------------------
__global__ __launch_bounds__(256) void k_head_swiglu(
    const float* __restrict__ pin, const float* __restrict__ normw,
    const float* __restrict__ gw, const float* __restrict__ gb,
    const float* __restrict__ vw, const float* __restrict__ vb,
    const float* __restrict__ ow, const float* __restrict__ ob,
    float* __restrict__ out) {
  const int h = blockIdx.x;
  __shared__ float P[B_][HD_];
  __shared__ __align__(16) _Float16 A[B_][HD_];
  __shared__ __align__(16) _Float16 Hact[B_][HQ_ + 8];
  __shared__ __align__(32) _Float16 WS[8][2048];  // wave-private swizzled tiles
  const int tid = threadIdx.x, w = tid >> 5, lane = tid & 31;

#pragma unroll
  for (int i = 0; i < (B_ * HD_) / 256; ++i) {
    int idx = tid + i * 256;
    int m = idx >> 7, d = idx & 127;
    P[m][d] = pin[((size_t)m * H_ + h) * HD_ + d];
  }
  __syncthreads();
#pragma unroll
  for (int j = 0; j < 2; ++j) {  // wave w: rows 2w, 2w+1
    int m = w * 2 + j;
    float ss = 0.f;
#pragma unroll
    for (int i = 0; i < 4; ++i) {
      float v = P[m][lane + i * 32];
      ss += v * v;
    }
    ss = wave_sum(ss);
    float rstd = rsqrtf(ss / (float)HD_ + EPS_);
#pragma unroll
    for (int i = 0; i < 4; ++i) {
      int d = lane + i * 32;
      A[m][d] = (_Float16)(P[m][d] * rstd * normw[d]);
    }
  }
  __syncthreads();

  const float* gwh = gw + (size_t)h * HD_ * HQ_;
  const float* vwh = vw + (size_t)h * HD_ * HQ_;
  const float* owh = ow + (size_t)h * HQ_ * HD_;
  _Float16* ws = WS[w];

  // GEMM1 gate: [16,128]x[128,512]; wave w owns n = w*64..w*64+63
  v8f ag[4] = {};
#pragma unroll
  for (int kk = 0; kk < HD_ / 32; ++kk) {
#pragma unroll
    for (int s = lane; s < 1024; s += 32) {
      int r = s >> 5, c = (s & 31) * 2;
      float2 g2 = *(const float2*)(gwh + (size_t)(kk * 32 + r) * HQ_ + w * 64 + c);
      _Float16* dst = swz_dst(ws, r, c);
      dst[0] = (_Float16)g2.x; dst[16] = (_Float16)g2.y;
    }
    v16h af = load_a_lds(&A[0][kk * 32], HD_);
#pragma unroll
    for (int j = 0; j < 4; ++j) ag[j] = wmma_f16(af, load_b_swz(ws, j), ag[j]);
  }
  // GEMM1 val
  v8f av[4] = {};
#pragma unroll
  for (int kk = 0; kk < HD_ / 32; ++kk) {
#pragma unroll
    for (int s = lane; s < 1024; s += 32) {
      int r = s >> 5, c = (s & 31) * 2;
      float2 v2 = *(const float2*)(vwh + (size_t)(kk * 32 + r) * HQ_ + w * 64 + c);
      _Float16* dst = swz_dst(ws, r, c);
      dst[0] = (_Float16)v2.x; dst[16] = (_Float16)v2.y;
    }
    v16h af = load_a_lds(&A[0][kk * 32], HD_);
#pragma unroll
    for (int j = 0; j < 4; ++j) av[j] = wmma_f16(af, load_b_swz(ws, j), av[j]);
  }

  const int col = lane & 15;
#pragma unroll
  for (int j = 0; j < 4; ++j) {
    int n = w * 64 + j * 16 + col;
    float gbias = gb[(size_t)h * HQ_ + n];
    float vbias = vb[(size_t)h * HQ_ + n];
#pragma unroll
    for (int r = 0; r < 8; ++r) {
      int m = (lane < 16) ? r : r + 8;
      float g = ag[j][r] + gbias, v = av[j][r] + vbias;
      Hact[m][n] = (_Float16)((g / (1.f + expf(-g))) * v);
    }
  }
  __syncthreads();

  // GEMM2: [16,512]x[512,128]; wave w owns n tile w*16
  v8f acc = {};
#pragma unroll
  for (int kk = 0; kk < HQ_ / 32; ++kk) {
#pragma unroll
    for (int s = lane; s < 256; s += 32) {
      int r = s >> 3, c = (s & 7) * 2;
      float2 o2 = *(const float2*)(owh + (size_t)(kk * 32 + r) * HD_ + w * 16 + c);
      _Float16* dst = swz_dst(ws, r, c);
      dst[0] = (_Float16)o2.x; dst[16] = (_Float16)o2.y;
    }
    v16h af = load_a_lds(&Hact[0][kk * 32], HQ_ + 8);
    acc = wmma_f16(af, load_b_swz(ws, 0), acc);
  }
  int n = w * 16 + col;
  float obias = ob[(size_t)h * HD_ + n];
#pragma unroll
  for (int r = 0; r < 8; ++r) {
    int m = (lane < 16) ? r : r + 8;
    out[((size_t)m * H_ + h) * HD_ + n] = acc[r] + obias + P[m][n];
  }
}

// ---------------------------------------------------------------------------
// Kernel 6: fused attention per (b,h) with folded K/V projections
// ---------------------------------------------------------------------------
__global__ __launch_bounds__(256) void k_attn(
    const float* __restrict__ qws, const float* __restrict__ seqh,
    const float* __restrict__ kw, const float* __restrict__ kb,
    const float* __restrict__ vw, const float* __restrict__ vb,
    const unsigned char* __restrict__ mask, float* __restrict__ zws) {
  const int b = blockIdx.x / H_, h = blockIdx.x % H_;
  __shared__ float qls[HD_], qk[HD_], sc[T_], ctx[HD_], red[256];
  const int tid = threadIdx.x, w = tid >> 5, lane = tid & 31;
  const float* qrow = qws + ((size_t)b * H_ + h) * HD_;
  if (tid < HD_) {
    qls[tid] = qrow[tid];
    ctx[tid] = 0.f;
  }
  __syncthreads();
  const float scale = rsqrtf((float)HD_);
  if (tid < HD_) {
    const float* kr = kw + ((size_t)h * HD_ + tid) * HD_;
    float s = 0.f;
    for (int e = 0; e < HD_; ++e) s += kr[e] * qls[e];
    qk[tid] = s * scale;
  }
  float pb = (tid < HD_) ? qls[tid] * kb[(size_t)h * HD_ + tid] : 0.f;
  red[tid] = pb;
  __syncthreads();
  for (int s = 128; s > 0; s >>= 1) {
    if (tid < s) red[tid] += red[tid + s];
    __syncthreads();
  }
  const float qb = red[0] * scale;
  __syncthreads();

  const float* base = seqh + ((size_t)b * T_) * D_ + (size_t)h * HD_;
  for (int t = w; t < T_; t += 8) {
    const float* hr = base + (size_t)t * D_;
    float4 hv = *(const float4*)(hr + lane * 4);
    float s = hv.x * qk[lane * 4] + hv.y * qk[lane * 4 + 1] +
              hv.z * qk[lane * 4 + 2] + hv.w * qk[lane * 4 + 3];
    s = wave_sum(s);
    if (lane == 0) {
      float sv = s + qb;
      if (!mask[(size_t)b * T_ + t]) sv = -INFINITY;
      sc[t] = sv;
    }
  }
  __syncthreads();

  float mx = -INFINITY;
  for (int i = tid; i < T_; i += 256) mx = fmaxf(mx, sc[i]);
  red[tid] = mx;
  __syncthreads();
  for (int s = 128; s > 0; s >>= 1) {
    if (tid < s) red[tid] = fmaxf(red[tid], red[tid + s]);
    __syncthreads();
  }
  mx = red[0];
  __syncthreads();
  float ps = 0.f;
  for (int i = tid; i < T_; i += 256) {
    float e = expf(sc[i] - mx);
    sc[i] = e;
    ps += e;
  }
  red[tid] = ps;
  __syncthreads();
  for (int s = 128; s > 0; s >>= 1) {
    if (tid < s) red[tid] += red[tid + s];
    __syncthreads();
  }
  const float inv = 1.f / red[0];
  __syncthreads();

  float a0 = 0.f, a1 = 0.f, a2 = 0.f, a3 = 0.f;
  for (int t = w * (T_ / 8); t < (w + 1) * (T_ / 8); ++t) {
    float a = sc[t] * inv;
    const float* hr = base + (size_t)t * D_;
    float4 hv = *(const float4*)(hr + lane * 4);
    a0 += a * hv.x; a1 += a * hv.y; a2 += a * hv.z; a3 += a * hv.w;
  }
  atomicAdd(&ctx[lane * 4 + 0], a0);
  atomicAdd(&ctx[lane * 4 + 1], a1);
  atomicAdd(&ctx[lane * 4 + 2], a2);
  atomicAdd(&ctx[lane * 4 + 3], a3);
  __syncthreads();

  if (tid < HD_) {
    const float* vwh = vw + (size_t)h * HD_ * HD_;
    float s = 0.f;
    for (int d = 0; d < HD_; ++d) s += ctx[d] * vwh[(size_t)d * HD_ + tid];
    zws[((size_t)b * H_ + h) * HD_ + tid] =
        s + vb[(size_t)h * HD_ + tid] + qls[tid];
  }
}

// ---------------------------------------------------------------------------
extern "C" void kernel_launch(void* const* d_in, const int* in_sizes, int n_in,
                              void* d_out, int out_size, void* d_ws,
                              size_t ws_size, hipStream_t stream) {
  const float* x_heads          = (const float*)d_in[0];
  const float* seq_repr         = (const float*)d_in[1];
  const unsigned char* seq_mask = (const unsigned char*)d_in[2];
  const float* qm_norm_in_w     = (const float*)d_in[3];
  const float* qm_norm_head_w   = (const float*)d_in[4];
  const float* qm_gate_w = (const float*)d_in[5];
  const float* qm_gate_b = (const float*)d_in[6];
  const float* qm_val_w  = (const float*)d_in[7];
  const float* qm_val_b  = (const float*)d_in[8];
  const float* qm_out_w  = (const float*)d_in[9];
  const float* qm_out_b  = (const float*)d_in[10];
  const float* sf_norm_w = (const float*)d_in[11];
  const float* sf_gate_w = (const float*)d_in[12];
  const float* sf_gate_b = (const float*)d_in[13];
  const float* sf_val_w  = (const float*)d_in[14];
  const float* sf_val_b  = (const float*)d_in[15];
  const float* sf_out_w  = (const float*)d_in[16];
  const float* sf_out_b  = (const float*)d_in[17];
  const float* k_w = (const float*)d_in[18];
  const float* k_b = (const float*)d_in[19];
  const float* v_w = (const float*)d_in[20];
  const float* v_b = (const float*)d_in[21];
  const float* of_norm_w = (const float*)d_in[22];
  const float* of_gate_w = (const float*)d_in[23];
  const float* of_gate_b = (const float*)d_in[24];
  const float* of_val_w  = (const float*)d_in[25];
  const float* of_val_b  = (const float*)d_in[26];
  const float* of_out_w  = (const float*)d_in[27];
  const float* of_out_b  = (const float*)d_in[28];

  float* o_out      = (float*)d_out;                          // [B,H,HD]
  float* seq_hidden = (float*)d_out + (size_t)B_ * H_ * HD_;  // [B,T,D]

  char* ws = (char*)d_ws;
  _Float16* xn   = (_Float16*)ws;                               // 64 MB
  _Float16* hbuf = (_Float16*)(ws + (size_t)B_ * T_ * D_ * 2);  // 256 MB
  float* pws = (float*)(ws + (size_t)B_ * T_ * D_ * 2 + (size_t)B_ * T_ * HF_ * 2);
  float* qws = pws + (size_t)B_ * H_ * HD_;
  float* zws = qws + (size_t)B_ * H_ * HD_;

  // --- QueryMixer ---
  k_qm_mix<<<B_, 256, 0, stream>>>(x_heads, qm_norm_in_w, pws);
  k_head_swiglu<<<H_, 256, 0, stream>>>(pws, qm_norm_head_w, qm_gate_w,
                                        qm_gate_b, qm_val_w, qm_val_b, qm_out_w,
                                        qm_out_b, qws);
  // --- Sequence FFN (dominant, memory-bound) ---
  k_seq_rmsnorm<<<B_ * T_, 256, 0, stream>>>(seq_repr, sf_norm_w, xn);
  dim3 g1(HF_ / NT, (B_ * T_) / MT);
  k_ffn_gateval<<<g1, 256, 0, stream>>>(xn, sf_gate_w, sf_gate_b, sf_val_w,
                                        sf_val_b, hbuf);
  dim3 g2(D_ / NT, (B_ * T_) / MT);
  k_ffn_out<<<g2, 256, 0, stream>>>(hbuf, sf_out_w, sf_out_b, seq_repr,
                                    seq_hidden);
  // --- Fused cross-attention (K/V projections folded) ---
  k_attn<<<B_ * H_, 256, 0, stream>>>(qws, seq_hidden, k_w, k_b, v_w, v_b,
                                      seq_mask, zws);
  // --- OutputFusion ---
  k_head_swiglu<<<H_, 256, 0, stream>>>(zws, of_norm_w, of_gate_w, of_gate_b,
                                        of_val_w, of_val_b, of_out_w, of_out_b,
                                        o_out);
}
